// DotProductAttention_60017872995131
// MI455X (gfx1250) — matrix operand order repaired and checked by reading, not compiled, not run on hardware
//
#include <hip/hip_runtime.h>

typedef __attribute__((ext_vector_type(16))) __bf16 v16bf;
typedef __attribute__((ext_vector_type(8)))  float  v8f;
typedef __attribute__((ext_vector_type(4)))  float  v4f;

#define BATCH   16
#define NQ      2048
#define NKV     2048
#define DH      64
#define QTILE   16
#define NWAVES  8
#define SROW    2052            // padded LDS row stride in floats (2048+4 -> bank skew)
#define OROW    68              // padded O-accum row stride (floats)
#define VTROW   136             // padded Vt row stride in bf16 elems (128 + 8)
#define KBLK    128             // keys per cooperative V block in phase 4
#define SCALE   0.125f          // 1/sqrt(64)
#define MASKVAL (-1.0e6f)

// floats before the Vt region
#define SMEM_FLOATS (QTILE * SROW + 16 + 16 + 256 + QTILE * OROW)
#define SMEM_BYTES  (SMEM_FLOATS * 4 + DH * VTROW * 2)

__device__ __forceinline__ void pack8(v16bf& d, int base, float4 x, float4 y) {
    d[base + 0] = (__bf16)x.x; d[base + 1] = (__bf16)x.y;
    d[base + 2] = (__bf16)x.z; d[base + 3] = (__bf16)x.w;
    d[base + 4] = (__bf16)y.x; d[base + 5] = (__bf16)y.y;
    d[base + 6] = (__bf16)y.z; d[base + 7] = (__bf16)y.w;
}

__global__ __launch_bounds__(256, 1)
void attn_fused_kernel(const float* __restrict__ Q,
                       const float* __restrict__ K,
                       const float* __restrict__ V,
                       const int*   __restrict__ vlen,
                       float* __restrict__ outO,   // [B, NQ, DH]
                       float* __restrict__ outA)   // [B, NQ, NKV]
{
    extern __shared__ char smem_raw[];
    float*  S      = (float*)smem_raw;            // [QTILE][SROW] scores -> exp values
    float*  rowMax = S + QTILE * SROW;            // [16]
    float*  rowInv = rowMax + 16;                 // [16]
    float*  red    = rowInv + 16;                 // [256]
    float*  Oacc   = red + 256;                   // [16][OROW]
    __bf16* Vt     = (__bf16*)(smem_raw + SMEM_FLOATS * 4);  // [DH][VTROW] transposed V block

    const int tid  = threadIdx.x;
    const int lane = tid & 31;
    const int wave = tid >> 5;
    const int m16  = lane & 15;                  // row / col-within-tile
    const int half = lane >> 4;                  // 0 or 1

    const int wg = blockIdx.x;
    const int b  = wg / (NQ / QTILE);
    const int q0 = (wg % (NQ / QTILE)) * QTILE;

    const int vl = vlen[b];

    const float* Qb = Q + ((size_t)b * NQ + q0) * DH;
    const float* Kb = K + (size_t)b * NKV * DH;
    const float* Vb = V + (size_t)b * NKV * DH;

    // ---- Load Q tile as two bf16 A-fragments (16x32), contiguous k layout:
    // lane holds M=lane&15; element s: k = half*16 + s
    v16bf a0, a1;
    {
        const float4* q4 = (const float4*)(Qb + (size_t)m16 * DH);
        pack8(a0, 0, q4[half * 4 + 0],     q4[half * 4 + 1]);      // k = half*16 + 0..7
        pack8(a0, 8, q4[half * 4 + 2], q4[half * 4 + 3]);  // k = half*16 + 8..15
        pack8(a1, 0, q4[8 + half * 4 + 0], q4[8 + half * 4 + 1]);  // +32
        pack8(a1, 8, q4[8 + half * 4 + 2], q4[8 + half * 4 + 3]);
    }

    // ---- Phase 1: S = (Q K^T) * scale, masked, into LDS ----------------------
    for (int it = 0; it < NKV / (16 * NWAVES); ++it) {
        const int kv0 = (it * NWAVES + wave) * 16;
        // B fragment (K^T, 32x16): lane col n=m16 -> key row kv0+m16;
        // k = half*16 + p*2 (+0/1) -> 16 consecutive features per lane.
        const float* krow = Kb + (size_t)(kv0 + m16) * DH + half * 16;
        __builtin_prefetch(krow + KBLK * DH, 0, 3);   // next tile for this wave
        const float4* k4 = (const float4*)krow;
        v16bf b0, b1;
        pack8(b0, 0, k4[0], k4[1]);
        pack8(b0, 8, k4[2], k4[3]);
        pack8(b1, 0, k4[8], k4[9]);       // +32 floats
        pack8(b1, 8, k4[10], k4[11]);
        v8f c = {};
        c = __builtin_amdgcn_wmma_f32_16x16x32_bf16(false, a0, false, b0, (short)0, c, false, false);
        c = __builtin_amdgcn_wmma_f32_16x16x32_bf16(false, a1, false, b1, (short)0, c, false, false);

        // C layout: element p -> M = p + half*8, N = m16 (global key col kv0+m16)
        const int col    = kv0 + m16;
        const bool valid = (col < vl);
        #pragma unroll
        for (int p = 0; p < 8; ++p) {
            float s = c[p] * SCALE;
            s = valid ? s : MASKVAL;
            S[(p + half * 8) * SROW + col] = s;
        }
    }
    __syncthreads();

    // ---- Phase 2: row max / exp / row sum (16 threads per row, float4) -------
    const int r   = tid >> 4;   // 0..15
    const int sub = tid & 15;
    {
        float mx = -3.0e38f;
        for (int i = 0; i < NKV / 64; ++i) {
            const float4 v = *(const float4*)&S[r * SROW + (sub + 16 * i) * 4];
            mx = fmaxf(mx, fmaxf(fmaxf(v.x, v.y), fmaxf(v.z, v.w)));
        }
        red[tid] = mx;
        __syncthreads();
        if (sub == 0) {
            float m2 = red[r * 16];
            #pragma unroll
            for (int j = 1; j < 16; ++j) m2 = fmaxf(m2, red[r * 16 + j]);
            rowMax[r] = m2;
        }
        __syncthreads();
        const float rm = rowMax[r];
        float sum = 0.f;
        for (int i = 0; i < NKV / 64; ++i) {
            float4* p4 = (float4*)&S[r * SROW + (sub + 16 * i) * 4];
            float4 v = *p4;
            v.x = __expf(v.x - rm); v.y = __expf(v.y - rm);
            v.z = __expf(v.z - rm); v.w = __expf(v.w - rm);
            sum += (v.x + v.y) + (v.z + v.w);
            *p4 = v;                       // keep un-normalized P in LDS
        }
        red[tid] = sum;
        __syncthreads();
        if (sub == 0) {
            float t = 0.f;
            #pragma unroll
            for (int j = 0; j < 16; ++j) t += red[r * 16 + j];
            rowInv[r] = 1.0f / t;
        }
        __syncthreads();
    }

    // ---- Phase 3: stream attn = P * invSum to HBM (non-temporal: 268MB once) -
    {
        const float inv = rowInv[r];
        v4f* arow = (v4f*)(outA + ((size_t)b * NQ + q0 + r) * NKV);
        for (int i = 0; i < 32; ++i) {
            const int c4 = sub + 16 * i;           // float4 index, coalesced
            const float* s4 = &S[r * SROW + c4 * 4];
            v4f v4;
            v4.x = s4[0] * inv; v4.y = s4[1] * inv;
            v4.z = s4[2] * inv; v4.w = s4[3] * inv;
            __builtin_nontemporal_store(v4, arow + c4);
        }
    }

    // ---- Phase 4: O = P . V, cooperative 128-key blocks ----------------------
    for (int idx = tid; idx < QTILE * OROW; idx += 256) Oacc[idx] = 0.f;

    v8f oc0 = {}, oc1 = {};
    const int id0   = wave * 2;
    const int id1   = wave * 2 + 1;
    for (int blk = 0; blk < NKV / KBLK; ++blk) {
        // Stage V block transposed to bf16: Vt[feature][key], coalesced b128 reads.
        {
            const float4* vsrc = (const float4*)(Vb + (size_t)blk * KBLK * DH);
            #pragma unroll
            for (int i = 0; i < (KBLK * DH / 4) / 256; ++i) {   // 8 float4s/thread
                const int f  = tid + 256 * i;
                const int kr = f >> 4;            // key row within block
                const int fb = (f & 15) * 4;      // feature base
                const float4 v = vsrc[f];
                Vt[(fb + 0) * VTROW + kr] = (__bf16)v.x;
                Vt[(fb + 1) * VTROW + kr] = (__bf16)v.y;
                Vt[(fb + 2) * VTROW + kr] = (__bf16)v.z;
                Vt[(fb + 3) * VTROW + kr] = (__bf16)v.w;
            }
        }
        __syncthreads();

        #pragma unroll
        for (int j = 0; j < 2; ++j) {
            const int id     = j ? id1 : id0;
            const int chunk  = id & 3;            // output column chunk
            const int klocal = (id >> 2) * 32;    // key offset within block
            const int k0     = blk * KBLK + klocal;

            // A fragment = P (bf16) from LDS exp values, float4 reads
            v16bf pa;
            {
                const float4* s4a = (const float4*)&S[m16 * SROW + k0 + half * 16];
                const float4* s4b = (const float4*)&S[m16 * SROW + k0 + half * 16 + 8];
                pack8(pa, 0, s4a[0], s4a[1]);
                pack8(pa, 8, s4b[0], s4b[1]);
            }
            // B fragment = one aligned 32B vector from transposed bf16 V tile:
            // element 2p+e <-> key klocal + half*16 + 2p + e  (contiguous order, same as A)
            const int n = chunk * 16 + m16;       // feature column
            const v16bf vbf = *(const v16bf*)(Vt + n * VTROW + klocal + half * 16);

            if (j == 0)
                oc0 = __builtin_amdgcn_wmma_f32_16x16x32_bf16(false, pa, false, vbf, (short)0, oc0, false, false);
            else
                oc1 = __builtin_amdgcn_wmma_f32_16x16x32_bf16(false, pa, false, vbf, (short)0, oc1, false, false);
        }
        __syncthreads();   // Vt reused next block
    }

    // ---- Cross-wave reduction via ds_add_f32 ---------------------------------
    #pragma unroll
    for (int j = 0; j < 2; ++j) {
        const int chunk = ((j ? id1 : id0) & 3);
        const int col   = chunk * 16 + m16;
        const v8f oc    = j ? oc1 : oc0;
        #pragma unroll
        for (int p = 0; p < 8; ++p)
            atomicAdd(&Oacc[(p + half * 8) * OROW + col], oc[p]);
    }
    __syncthreads();

    // ---- Final out store: 1024 floats, float4 per thread, scaled by 1/rowsum -
    {
        const int idx = tid * 4;          // 0..1023
        const int m   = idx >> 6;         // /DH
        const int c0  = idx & 63;
        const float inv = rowInv[m];
        const float* oa = &Oacc[m * OROW + c0];
        float4 o4;
        o4.x = oa[0] * inv; o4.y = oa[1] * inv;
        o4.z = oa[2] * inv; o4.w = oa[3] * inv;
        float* orow = outO + ((size_t)b * NQ + q0 + m) * DH;
        *(float4*)(orow + c0) = o4;
    }
}

extern "C" void kernel_launch(void* const* d_in, const int* in_sizes, int n_in,
                              void* d_out, int out_size, void* d_ws, size_t ws_size,
                              hipStream_t stream) {
    const float* Q  = (const float*)d_in[0];
    const float* K  = (const float*)d_in[1];
    const float* V  = (const float*)d_in[2];
    const int*   vl = (const int*)d_in[3];

    float* outO = (float*)d_out;                               // [B,NQ,DH]
    float* outA = outO + (size_t)BATCH * NQ * DH;              // [B,NQ,NKV]

    // ~154 KB dynamic LDS (WGP has 320KB -> 2 workgroups per WGP).
    (void)hipFuncSetAttribute((const void*)attn_fused_kernel,
                              hipFuncAttributeMaxDynamicSharedMemorySize,
                              (int)SMEM_BYTES);

    dim3 grid(BATCH * (NQ / QTILE));   // 2048 workgroups
    attn_fused_kernel<<<grid, 256, (size_t)SMEM_BYTES, stream>>>(Q, K, V, vl, outO, outA);
}
